// EdgeConvMod_24833500905986
// MI455X (gfx1250) — matrix operand order repaired and checked by reading, not compiled, not run on hardware
//
#include <hip/hip_runtime.h>

// ---- problem constants (match reference) ----
#define NN   50000
#define EE   800000
#define DD   64
#define DEE  32
#define HH   256
#define FI   160          // 2*D + DE
#define TILE_E 64         // edges per block
#define THREADS 128       // 4 wave32 waves
#define FSTR 168          // padded feat row stride (bf16 elems), 336B = 21*16B, conflict-free
#define HSTR 264          // padded h row stride (bf16 elems), 528B = 33*16B

typedef __attribute__((ext_vector_type(16))) __bf16 v16bf;
typedef __attribute__((ext_vector_type(8)))  float  v8f;

union BFrag { uint4 u[2]; v16bf v; };

__device__ __forceinline__ unsigned short f2bf(float f) {
    unsigned int u = __float_as_uint(f);
    u += 0x7FFFu + ((u >> 16) & 1u);   // round-to-nearest-even
    return (unsigned short)(u >> 16);
}

// ordered-float atomic max via int-max / uint-min trick (init must be -inf bits)
__device__ __forceinline__ void atomic_max_f32(float* addr, float v) {
    if (v >= 0.0f) atomicMax((int*)addr, __float_as_int(v));
    else           atomicMin((unsigned int*)addr, __float_as_uint(v));
}

// ---------------- weight repack: fp32 row-major -> bf16 WMMA B-fragment tiles ----------------
// tile = (kt, nt): 32(K) x 16(N); storage: tile*512 + lane*16 + e  (bf16 elems)
// lane L: N = L%16 ; K = kt*32 + (L/16)*8 + (e%8) + (e/8)*16
__global__ void pack_w1_kernel(const float* __restrict__ W, unsigned short* __restrict__ Wp) {
    int idx = blockIdx.x * 256 + threadIdx.x;          // < 160*256
    int tile = idx >> 9, within = idx & 511;
    int lane = within >> 4, e = within & 15;
    int kt = tile >> 4, nt = tile & 15;                // 16 N-tiles for H=256
    int k = kt * 32 + (lane >> 4) * 8 + (e & 7) + (e >> 3) * 16;
    int n = nt * 16 + (lane & 15);
    Wp[idx] = f2bf(W[(size_t)k * HH + n]);
}

__global__ void pack_w2_kernel(const float* __restrict__ W, unsigned short* __restrict__ Wp) {
    int idx = blockIdx.x * 256 + threadIdx.x;          // < 256*64
    int tile = idx >> 9, within = idx & 511;
    int lane = within >> 4, e = within & 15;
    int kt = tile >> 2, nt = tile & 3;                 // 4 N-tiles for D=64
    int k = kt * 32 + (lane >> 4) * 8 + (e & 7) + (e >> 3) * 16;
    int n = nt * 16 + (lane & 15);
    Wp[idx] = f2bf(W[(size_t)k * DD + n]);
}

__global__ void init_agg_kernel(unsigned int* __restrict__ agg) {
    int i = blockIdx.x * 256 + threadIdx.x;
    agg[i] = 0xFF800000u;                               // -inf bits
}

__global__ void fixup_agg_kernel(float* __restrict__ agg) {
    int i = blockIdx.x * 256 + threadIdx.x;
    if (__float_as_uint(agg[i]) == 0xFF800000u) agg[i] = 0.0f;
}

// ---------------- main fused kernel ----------------
__global__ __launch_bounds__(THREADS) void edgeconv_main_kernel(
    const float* __restrict__ x, const int* __restrict__ ei,
    const float* __restrict__ ea,
    const unsigned short* __restrict__ W1p, const float* __restrict__ b1,
    const unsigned short* __restrict__ W2p, const float* __restrict__ b2,
    float* __restrict__ agg, float* __restrict__ outp)
{
    __shared__ unsigned short s_feat[TILE_E * FSTR];   // 21.0 KB
    __shared__ unsigned short s_h[TILE_E * HSTR];      // 33.0 KB
    __shared__ int s_row[TILE_E];
    __shared__ int s_col[TILE_E];

    const int tid  = threadIdx.x;
    const int lane = tid & 31;
    const int w    = tid >> 5;        // wave id 0..3
    const int grp  = lane >> 4;       // lane group 0/1
    const int nl   = lane & 15;
    const int ebase = blockIdx.x * TILE_E;

    if (tid < TILE_E)           s_row[tid] = ei[ebase + tid];
    else                        s_col[tid - TILE_E] = ei[EE + ebase + (tid - TILE_E)];
    __syncthreads();

    // ---- gather feats = [x[row] | x[col] | edge_attr] -> bf16 LDS tile (float4 vectorized)
    for (int i = tid; i < TILE_E * (FI / 4); i += THREADS) {
        int e = i / (FI / 4);
        int q = i % (FI / 4);
        float4 v;
        if (q < 16)      v = *(const float4*)(x  + (size_t)s_row[e] * DD  + q * 4);
        else if (q < 32) v = *(const float4*)(x  + (size_t)s_col[e] * DD  + (q - 16) * 4);
        else             v = *(const float4*)(ea + (size_t)(ebase + e) * DEE + (q - 32) * 4);
        ushort4 p;
        p.x = f2bf(v.x); p.y = f2bf(v.y); p.z = f2bf(v.z); p.w = f2bf(v.w);
        *(ushort4*)(s_feat + e * FSTR + q * 4) = p;
    }
    __syncthreads();

    // ---- GEMM1: [64 x 160] @ [160 x 256], wave w owns cols [w*64, w*64+64)
    v8f vzero = {};
    v8f acc[16];
    #pragma unroll
    for (int a = 0; a < 16; ++a) acc[a] = vzero;

    for (int kt = 0; kt < FI / 32; ++kt) {
        if (kt + 1 < FI / 32)   // prefetch next K-step's B tiles (global_prefetch_b8)
            __builtin_prefetch(W1p + (size_t)((kt + 1) * 16 + w * 4) * 512 + lane * 16, 0, 1);

        BFrag bf[4];
        #pragma unroll
        for (int nt = 0; nt < 4; ++nt) {
            const uint4* src = (const uint4*)(W1p + (size_t)(kt * 16 + w * 4 + nt) * 512 + lane * 16);
            bf[nt].u[0] = src[0];
            bf[nt].u[1] = src[1];
        }
        #pragma unroll
        for (int mt = 0; mt < 4; ++mt) {
            BFrag af;   // A frag: row = mt*16 + nl, K chunks [8*grp..8*grp+7] and +16
            const unsigned short* ap = s_feat + (mt * 16 + nl) * FSTR + kt * 32 + 8 * grp;
            af.u[0] = *(const uint4*)(ap);
            af.u[1] = *(const uint4*)(ap + 16);
            #pragma unroll
            for (int nt = 0; nt < 4; ++nt) {
                acc[mt * 4 + nt] = __builtin_amdgcn_wmma_f32_16x16x32_bf16(
                    false, af.v, false, bf[nt].v, (short)0, acc[mt * 4 + nt], false, false);
            }
        }
    }

    // ---- bias + ReLU -> bf16 h tile in LDS (lane-layout reshuffle for GEMM2 A)
    #pragma unroll
    for (int nt = 0; nt < 4; ++nt) {
        int ncol = (w * 4 + nt) * 16 + nl;
        float bias = b1[ncol];
        #pragma unroll
        for (int mt = 0; mt < 4; ++mt) {
            #pragma unroll
            for (int r = 0; r < 8; ++r) {
                int rowl = mt * 16 + r + 8 * grp;
                float vv = acc[mt * 4 + nt][r] + bias;
                vv = vv > 0.0f ? vv : 0.0f;
                s_h[rowl * HSTR + ncol] = f2bf(vv);
            }
        }
    }
    __syncthreads();

    // ---- GEMM2: [64 x 256] @ [256 x 64], wave w owns rows [w*16, w*16+16)
    v8f acc2[4];
    #pragma unroll
    for (int a = 0; a < 4; ++a) acc2[a] = vzero;

    for (int kt = 0; kt < HH / 32; ++kt) {
        if (kt + 1 < HH / 32)
            __builtin_prefetch(W2p + (size_t)((kt + 1) * 4) * 512 + lane * 16, 0, 1);

        BFrag af;
        const unsigned short* ap = s_h + (w * 16 + nl) * HSTR + kt * 32 + 8 * grp;
        af.u[0] = *(const uint4*)(ap);
        af.u[1] = *(const uint4*)(ap + 16);
        #pragma unroll
        for (int nt = 0; nt < 4; ++nt) {
            BFrag bfr;
            const uint4* src = (const uint4*)(W2p + (size_t)(kt * 4 + nt) * 512 + lane * 16);
            bfr.u[0] = src[0];
            bfr.u[1] = src[1];
            acc2[nt] = __builtin_amdgcn_wmma_f32_16x16x32_bf16(
                false, af.v, false, bfr.v, (short)0, acc2[nt], false, false);
        }
    }

    // ---- epilogue: bias, store out (fp32), scatter-max into agg
    #pragma unroll
    for (int nt = 0; nt < 4; ++nt) {
        int colg = nt * 16 + nl;
        float bias = b2[colg];
        #pragma unroll
        for (int r = 0; r < 8; ++r) {
            int rowl = w * 16 + r + 8 * grp;
            float vv = acc2[nt][r] + bias;
            outp[(size_t)(ebase + rowl) * DD + colg] = vv;
            atomic_max_f32(agg + (size_t)s_row[rowl] * DD + colg, vv);
        }
    }
}

extern "C" void kernel_launch(void* const* d_in, const int* in_sizes, int n_in,
                              void* d_out, int out_size, void* d_ws, size_t ws_size,
                              hipStream_t stream) {
    (void)in_sizes; (void)n_in; (void)out_size; (void)ws_size;
    const float* x  = (const float*)d_in[0];
    const int*   ei = (const int*)d_in[1];
    const float* ea = (const float*)d_in[2];
    const float* W1 = (const float*)d_in[3];
    const float* b1 = (const float*)d_in[4];
    const float* W2 = (const float*)d_in[5];
    const float* b2 = (const float*)d_in[6];

    float* agg  = (float*)d_out;                       // [N, D]
    float* outp = agg + (size_t)NN * DD;               // [E, D]

    unsigned short* W1p = (unsigned short*)d_ws;       // 160*256 bf16
    unsigned short* W2p = W1p + (size_t)FI * HH;       // 256*64 bf16

    pack_w1_kernel<<<(FI * HH) / 256, 256, 0, stream>>>(W1, W1p);
    pack_w2_kernel<<<(HH * DD) / 256, 256, 0, stream>>>(W2, W2p);
    init_agg_kernel<<<((size_t)NN * DD) / 256, 256, 0, stream>>>((unsigned int*)agg);

    edgeconv_main_kernel<<<EE / TILE_E, THREADS, 0, stream>>>(
        x, ei, ea, W1p, b1, W2p, b2, agg, outp);

    fixup_agg_kernel<<<((size_t)NN * DD) / 256, 256, 0, stream>>>(agg);
}
